// ScaledDotProductAttention_77781857731063
// MI455X (gfx1250) — matrix operand order repaired and checked by reading, not compiled, not run on hardware
//
#include <hip/hip_runtime.h>

// Problem constants (fixed by the reference).
#define SB   2
#define SH   16
#define SS   2048
#define SDK  64
#define SPITCH 2052   // score row pitch in floats (2052 % 64 == 4 -> bank-skewed; % 4 == 0 -> vec4 ok)

typedef __attribute__((ext_vector_type(16))) __bf16 v16bf;
typedef __attribute__((ext_vector_type(8)))  float  v8f;
typedef __attribute__((ext_vector_type(4)))  float  v4f;   // native vector: b128 + NT-store capable

// Native RNE convert; backend should lower to hardware bf16 cvt on gfx1250.
static __device__ inline __bf16 cvt_bf16(float f) { return (__bf16)f; }

__global__ __launch_bounds__(128) void
attn_wmma_kernel(const float* __restrict__ Q, const float* __restrict__ K,
                 const float* __restrict__ V, const int* __restrict__ mask,
                 const float* __restrict__ seq_mask,
                 float* __restrict__ out, float* __restrict__ attn) {
    // LDS carve-out (dynamic).
    extern __shared__ float smemf[];
    float* seqm    = smemf;             // [SS]
    float* sc      = seqm + SS;         // [16][SPITCH] scores -> probs
    float* red     = sc + 16 * SPITCH;  // [128]
    float* red2    = red + 128;         // [128]
    float* rowmax  = red2 + 128;        // [16]
    float* rowsum  = rowmax + 16;       // [16]
    float* rowsumm = rowsum + 16;       // [16]

    const int bh    = blockIdx.z * SH + blockIdx.y;      // fused (b,h)
    const int qbase = blockIdx.x * 16;                   // query tile base
    const int tid   = threadIdx.x;
    const int lane  = tid & 31;
    const int wave  = tid >> 5;                          // 0..3
    const int m     = lane & 15;                         // M row / N col id
    const int hi    = lane >> 4;                         // lane half

    // Preload per-key seq mask into LDS (coalesced, b128).
    {
        const v4f* g4 = (const v4f*)seq_mask;
        v4f*       s4 = (v4f*)seqm;
        for (int k4 = tid; k4 < SS / 4; k4 += 128) s4[k4] = g4[k4];
    }

    // ---- Phase 1: scores = (Q K^T) * 1/sqrt(64), masked, into LDS ----
    // Fold the 0.125 scale into A: power-of-two scale is exact in fp32 and bf16.
    v16bf aq[2];
    {
        const float* qrow = Q + ((size_t)bh * SS + qbase + m) * SDK;
        #pragma unroll
        for (int c = 0; c < 2; ++c) {
            const int dbase = 32 * c;
            #pragma unroll
            for (int e = 0; e < 16; ++e) {
                const int kc = dbase + ((e < 8) ? 0 : 16) + hi * 8 + (e & 7);
                aq[c][e] = cvt_bf16(qrow[kc] * 0.125f);
            }
        }
    }

    for (int kt = wave; kt < SS / 16; kt += 4) {
        const int kbase = kt * 16;
        // B fragments: B[kc][n] = Key[kbase+n][kc]; lane n = m, two 8-float runs per 32-k chunk.
        const float* krow = K + ((size_t)bh * SS + kbase + m) * SDK;
        v16bf bk[2];
        #pragma unroll
        for (int c = 0; c < 2; ++c) {
            const int dbase = 32 * c;
            #pragma unroll
            for (int e = 0; e < 16; ++e) bk[c][e] = cvt_bf16(krow[dbase + ((e < 8) ? 0 : 16) + hi * 8 + (e & 7)]);
        }
        v8f acc = {};
        acc = __builtin_amdgcn_wmma_f32_16x16x32_bf16(false, aq[0], false, bk[0],
                                                      (short)0, acc, false, false);
        acc = __builtin_amdgcn_wmma_f32_16x16x32_bf16(false, aq[1], false, bk[1],
                                                      (short)0, acc, false, false);
        // Mask + store tile to LDS.  C/D layout: element r -> M = r + 8*hi, N = m.
        #pragma unroll
        for (int r = 0; r < 8; ++r) {
            const int mr = r + 8 * hi;
            float s = acc[r];
            const int q = qbase + mr;
            if (mask[(size_t)q * SS + kbase + m] == 0) s = -1e9f;
            sc[mr * SPITCH + kbase + m] = s;
        }
    }
    __syncthreads();

    // ---- Phase 2: per-row max, sum(exp), sum(exp*seqmask)  (vec4 sweeps) ----
    {
        const int r    = tid >> 3;   // 0..15
        const int part = tid & 7;    // 0..7
        const v4f* srow = (const v4f*)&sc[r * SPITCH + part * 256];
        float mx = -3.4e38f;
        for (int j = 0; j < 64; ++j) {
            const v4f v = srow[j];
            mx = fmaxf(mx, fmaxf(fmaxf(v.x, v.y), fmaxf(v.z, v.w)));
        }
        red[r * 8 + part] = mx;
        __syncthreads();
        if (part == 0) {
            float v = red[r * 8];
            for (int i = 1; i < 8; ++i) v = fmaxf(v, red[r * 8 + i]);
            rowmax[r] = v;
        }
        __syncthreads();
        const float rmax = rowmax[r];
        const v4f* wrow = (const v4f*)&seqm[part * 256];
        float se = 0.f, sm = 0.f;
        for (int j = 0; j < 64; ++j) {
            const v4f v = srow[j];
            const v4f w = wrow[j];
            const float e0 = __expf(v.x - rmax), e1 = __expf(v.y - rmax);
            const float e2 = __expf(v.z - rmax), e3 = __expf(v.w - rmax);
            se += (e0 + e1) + (e2 + e3);
            sm += (e0 * w.x + e1 * w.y) + (e2 * w.z + e3 * w.w);
        }
        red[r * 8 + part]  = se;
        red2[r * 8 + part] = sm;
        __syncthreads();
        if (part == 0) {
            float a = 0.f, b = 0.f;
            for (int i = 0; i < 8; ++i) { a += red[r * 8 + i]; b += red2[r * 8 + i]; }
            rowsum[r]  = a;
            rowsumm[r] = b;
        }
        __syncthreads();
    }

    // ---- Phase 3: normalized probs -> global attention (NT b128 streams) + LDS ----
    // attention = exp * seqm / max(sum_masked, sum_exp * 1e-20)  (matches ref clip)
    for (int rr = 0; rr < 16; ++rr) {
        const float rmax = rowmax[rr];
        const float inv  = 1.0f / fmaxf(rowsumm[rr], rowsum[rr] * 1e-20f);
        v4f*       prow  = (v4f*)&sc[rr * SPITCH];
        v4f*       arow4 = (v4f*)(attn + ((size_t)bh * SS + qbase + rr) * SS);
        const v4f* wm4   = (const v4f*)seqm;
        for (int k4 = tid; k4 < SS / 4; k4 += 128) {
            const v4f s = prow[k4];
            const v4f w = wm4[k4];
            v4f p;
            p.x = __expf(s.x - rmax) * w.x * inv;
            p.y = __expf(s.y - rmax) * w.y * inv;
            p.z = __expf(s.z - rmax) * w.z * inv;
            p.w = __expf(s.w - rmax) * w.w * inv;
            __builtin_nontemporal_store(p, &arow4[k4]);  // written once, never re-read
            prow[k4] = p;                                // reuse LDS as P for phase 4
        }
    }
    __syncthreads();

    // ---- Phase 4: out = P (16 x 2048) * V (2048 x 64), wave owns one 16-col tile ----
    const int ntile = wave;            // DK column tile, 4 tiles of 16
    v8f oacc = {};
    const float* vcol = V + (size_t)bh * SS * SDK + ntile * 16 + m;
    for (int kk = 0; kk < SS / 32; ++kk) {
        const int kb = kk * 32;
        v16bf ap, bv;
        #pragma unroll
        for (int e = 0; e < 16; ++e) {
            const int kc = kb + ((e < 8) ? 0 : 16) + hi * 8 + (e & 7);
            ap[e] = cvt_bf16(sc[m * SPITCH + kc]);
        }
        #pragma unroll
        for (int e = 0; e < 16; ++e) {
            const int key = kb + ((e < 8) ? 0 : 16) + hi * 8 + (e & 7);
            bv[e] = cvt_bf16(vcol[(size_t)key * SDK]);   // coalesced across lanes (n)
        }
        oacc = __builtin_amdgcn_wmma_f32_16x16x32_bf16(false, ap, false, bv,
                                                       (short)0, oacc, false, false);
    }
    #pragma unroll
    for (int r = 0; r < 8; ++r) {
        const int mr = r + 8 * hi;
        out[((size_t)bh * SS + qbase + mr) * SDK + ntile * 16 + m] = oacc[r];
    }
}

extern "C" void kernel_launch(void* const* d_in, const int* in_sizes, int n_in,
                              void* d_out, int out_size, void* d_ws, size_t ws_size,
                              hipStream_t stream) {
    (void)in_sizes; (void)n_in; (void)d_ws; (void)ws_size; (void)out_size;
    const float* Q    = (const float*)d_in[0];
    const float* K    = (const float*)d_in[1];
    const float* V    = (const float*)d_in[2];
    const int*   mask = (const int*)d_in[3];
    const float* seqm = (const float*)d_in[4];

    float* out  = (float*)d_out;                         // [B,H,S,DK] first
    float* attn = out + (size_t)SB * SH * SS * SDK;      // then [B,H,S,S]

    const size_t smem_bytes =
        (size_t)(SS + 16 * SPITCH + 128 + 128 + 16 + 16 + 16) * sizeof(float);

    // Opt in to >64KB dynamic LDS (CDNA5 WGP has 320KB; 2 blocks/WGP still fit).
    (void)hipFuncSetAttribute((const void*)attn_wmma_kernel,
                              hipFuncAttributeMaxDynamicSharedMemorySize,
                              (int)smem_bytes);

    dim3 grid(SS / 16, SH, SB);   // (query tiles, H, B)
    dim3 block(128);              // 4 wave32 waves
    attn_wmma_kernel<<<grid, block, smem_bytes, stream>>>(Q, K, V, mask, seqm, out, attn);
}
